// GridConv_4629974745453
// MI455X (gfx1250) — matrix operand order repaired and checked by reading, not compiled, not run on hardware
//
#include <hip/hip_runtime.h>
#include <hip/hip_bf16.h>
#include <stdint.h>

// Problem constants (match reference)
#define Bv    4
#define Kv    256
#define Nv    1024
#define Cv    256
#define GPv   64
#define NQv   (Bv * Kv * GPv)   // 65536 query points
#define CIN0  259               // 3 + 256
#define CINP0 288               // pad to multiple of 32
#define MPc   128
#define OUT3v 77
#define MP3   80                // 77 padded to multiple of 16
#define BKv   (Bv * Kv)         // 1024
#define EPSBN 1e-5f

#define NBLK  128               // output columns (queries) per workgroup
#define KCH   128               // K-chunk staged in LDS (32 KB tile)

typedef __attribute__((ext_vector_type(16))) __bf16 v16bf;
typedef __attribute__((ext_vector_type(8)))  float  v8f;

// CDNA5 async global->LDS path (ASYNCcnt-tracked), guarded so either toolchain compiles.
#if defined(__gfx1250__) && \
    __has_builtin(__builtin_amdgcn_global_load_async_to_lds_b128) && \
    __has_builtin(__builtin_amdgcn_s_wait_asynccnt)
#define USE_ASYNC_LDS 1
#else
#define USE_ASYNC_LDS 0
#endif

#if USE_ASYNC_LDS
typedef __attribute__((__vector_size__(4 * sizeof(int)))) int v4i_t;
__device__ __forceinline__ void async_copy16(const void* gsrc, void* ldst) {
    // signature: (v4i as1* src, v4i as3* dst, imm offset, imm cpol)
    __builtin_amdgcn_global_load_async_to_lds_b128(
        (__attribute__((address_space(1))) v4i_t*)gsrc,
        (__attribute__((address_space(3))) v4i_t*)ldst,
        0, 0);
}
#endif

#if __has_builtin(__builtin_amdgcn_sched_barrier)
#define SCHED_FENCE() __builtin_amdgcn_sched_barrier(0)
#else
#define SCHED_FENCE()
#endif

__device__ __forceinline__ unsigned short f2bf(float f) {
    uint32_t u = __float_as_uint(f);
    u += 0x7FFFu + ((u >> 16) & 1u);          // round to nearest even
    return (unsigned short)(u >> 16);
}
__device__ __forceinline__ float bf2f(unsigned short h) {
    return __uint_as_float(((uint32_t)h) << 16);
}

// ---------------------------------------------------------------------------
// Kernel 1: grid generation + 3-NN + inverse-distance interpolation.
// One block per (b,k) proposal; seeds staged in LDS.
// ---------------------------------------------------------------------------
__global__ void knn_interp_kernel(const float* __restrict__ center,
                                  const float* __restrict__ size,
                                  const float* __restrict__ heading,
                                  const float* __restrict__ seed_xyz,
                                  const float* __restrict__ seed_feat,
                                  unsigned short* __restrict__ x0) {
    __shared__ float sx[Nv], sy[Nv], sz[Nv];
    __shared__ int   sidx[GPv][3];
    __shared__ float sw[GPv][3];

    const int k = blockIdx.x, b = blockIdx.y;
    const int t = threadIdx.x;

    const float* sb = seed_xyz + (size_t)b * Nv * 3;
    for (int i = t; i < Nv; i += blockDim.x) {
        sx[i] = sb[i * 3 + 0];
        sy[i] = sb[i * 3 + 1];
        sz[i] = sb[i * 3 + 2];
    }
    __syncthreads();

    const size_t bk = (size_t)b * Kv + k;

    if (t < GPv) {
        // 4x4x4 grid, x varies slowest
        const int ix = t >> 4, iy = (t >> 2) & 3, iz = t & 3;
        const float stepv = 2.0f / 3.0f;
        const float gx = -1.0f + stepv * ix;
        const float gy = -1.0f + stepv * iy;
        const float gz = -1.0f + stepv * iz;

        const float szx = size[bk * 3 + 0], szy = size[bk * 3 + 1], szz = size[bk * 3 + 2];
        const float h  = heading[bk];
        const float c_ = cosf(h), s_ = sinf(h);
        const float px = gx * szx, py = gy * szy, pz = gz * szz;
        const float rx = px * c_ - py * s_;
        const float ry = px * s_ + py * c_;
        const float rz = pz;
        const float wx = rx + center[bk * 3 + 0];
        const float wy = ry + center[bk * 3 + 1];
        const float wz = rz + center[bk * 3 + 2];

        // top-3 smallest squared distances
        float d0 = 3.4e38f, d1 = 3.4e38f, d2q = 3.4e38f;
        int   i0 = 0, i1 = 0, i2 = 0;
        for (int n = 0; n < Nv; ++n) {
            const float dx = wx - sx[n], dy = wy - sy[n], dz = wz - sz[n];
            const float d = dx * dx + dy * dy + dz * dz;
            if (d < d0)      { d2q = d1; i2 = i1; d1 = d0; i1 = i0; d0 = d; i0 = n; }
            else if (d < d1) { d2q = d1; i2 = i1; d1 = d;  i1 = n; }
            else if (d < d2q){ d2q = d;  i2 = n; }
        }
        float w0 = 1.0f / (sqrtf(fmaxf(d0,  0.f)) + 1e-8f);
        float w1 = 1.0f / (sqrtf(fmaxf(d1,  0.f)) + 1e-8f);
        float w2 = 1.0f / (sqrtf(fmaxf(d2q, 0.f)) + 1e-8f);
        const float wsum = w0 + w1 + w2;
        sidx[t][0] = i0; sidx[t][1] = i1; sidx[t][2] = i2;
        sw[t][0] = w0 / wsum; sw[t][1] = w1 / wsum; sw[t][2] = w2 / wsum;

        unsigned short* row = x0 + (bk * GPv + t) * CINP0;
        row[0] = f2bf(rx);
        row[1] = f2bf(ry);
        row[2] = f2bf(rz);
        for (int c = 3 + Cv; c < CINP0; ++c) row[c] = 0;   // zero K-pad
    }
    __syncthreads();

    // interpolation: thread = feature channel (blockDim == Cv == 256)
    {
        const int c = t;
        const float* fb = seed_feat + ((size_t)b * Cv + c) * Nv;
        for (int g = 0; g < GPv; ++g) {
            const float v = sw[g][0] * fb[sidx[g][0]]
                          + sw[g][1] * fb[sidx[g][1]]
                          + sw[g][2] * fb[sidx[g][2]];
            x0[(bk * GPv + g) * CINP0 + 3 + c] = f2bf(v);
        }
    }
}

// ---------------------------------------------------------------------------
// Kernel 2: convert f32 weights [Cout][Cin] -> zero-padded bf16 [MPl][KP]
// ---------------------------------------------------------------------------
__global__ void cvt_w_kernel(const float* __restrict__ Wf, unsigned short* __restrict__ Wb,
                             int Cout, int Cin, int MPl, int KP) {
    const int idx = blockIdx.x * blockDim.x + threadIdx.x;
    if (idx >= MPl * KP) return;
    const int m = idx / KP, kq = idx % KP;
    const float v = (m < Cout && kq < Cin) ? Wf[(size_t)m * Cin + kq] : 0.0f;
    Wb[idx] = f2bf(v);
}

__global__ void pad_bias_kernel(const float* __restrict__ src, float* __restrict__ dst,
                                int nsrc, int ndst) {
    const int i = blockIdx.x * blockDim.x + threadIdx.x;
    if (i >= ndst) return;
    dst[i] = (i < nsrc) ? src[i] : 0.0f;
}

// ---------------------------------------------------------------------------
// Kernel 3: tiled bf16 WMMA GEMM.  Y[n][m] = sum_k W[m][k] * X[n][k] + bias[m]
//   Block = 8 waves (256 thr), output tile = [MP rows] x [NBLK=128 cols].
//   Wave w owns M-tile w; per K-step the A (weight) fragment is loaded once
//   (2x global b128) and reused for 8 N-subtiles.  The 128-column activation
//   slab is staged in LDS in 32KB K-chunks via CDNA5 async global->LDS loads
//   (ASYNCcnt) when available.  All 8 B fragments are prefetched from LDS
//   (16x ds_load_b128); a sched_barrier pins the grouping so the 8 WMMAs
//   issue back-to-back behind a single dscnt wait.
// Fragment lane layouts per CDNA5 ISA:
//   A 16x32: m=lane&15; lanes<16: K=0..7,16..23, lanes>=16: K=8..15,24..31
//   B 32x16: n=lane&15; lanes<16: K=0..7,16..23, lanes>=16: K=8..15,24..31
//   D 16x16: VGPR v <-> M = v + 8*(lane>=16), N = lane&15
// ---------------------------------------------------------------------------
__global__ __launch_bounds__(256)
void wmma_gemm_bf16_kernel(const unsigned short* __restrict__ W,
                           const unsigned short* __restrict__ X,
                           const float* __restrict__ bias,
                           float* __restrict__ Y,
                           int MP, int KP) {
    __shared__ unsigned short sX[NBLK * KCH];   // 32 KB activation slab

    const int tid  = threadIdx.x;
    const int wave = tid >> 5;
    const int lane = tid & 31;
    const int row  = lane & 15;
    const int hi   = lane >> 4;            // 0 or 1
    const int n0   = blockIdx.x * NBLK;    // first output column of this block
    const int mtiles = MP >> 4;
    const bool active = wave < mtiles;
    const int mt = active ? wave : 0;

    union U { v16bf v; uint32_t u[8]; uint4 q[2]; };
    v8f acc[8];
#pragma unroll
    for (int i = 0; i < 8; ++i) acc[i] = (v8f){};

    const unsigned short* wrow = W + (size_t)(mt * 16 + row) * KP;

    for (int k0 = 0; k0 < KP; k0 += KCH) {
        const int kc = (KP - k0 < KCH) ? (KP - k0) : KCH;
        const int rowBytes = kc * 2;                 // bytes per staged row
        const int perRow   = rowBytes >> 4;          // 16-byte chunks per row
        const int chunks   = NBLK * perRow;

        for (int ch = tid; ch < chunks; ch += 256) {
            const int r     = ch / perRow;
            const int cbyte = (ch - r * perRow) << 4;
            const unsigned short* gsrc = X + (size_t)(n0 + r) * KP + k0 + (cbyte >> 1);
            unsigned short*       ldst = sX + r * kc + (cbyte >> 1);
#if USE_ASYNC_LDS
            async_copy16(gsrc, ldst);
#else
            *(uint4*)ldst = *(const uint4*)gsrc;
#endif
        }
#if USE_ASYNC_LDS
        __builtin_amdgcn_s_wait_asynccnt(0);
#endif
        __syncthreads();

        if (active) {
            for (int kk = 0; kk < kc; kk += 32) {
                // A fragment: two 16B chunks at k-offsets {0..7} and {16..23} (+hi*8)
                U a;
                const uint4* wp4 = (const uint4*)(wrow + k0 + kk + hi * 8);
                a.q[0] = wp4[0];
                a.q[1] = wp4[2];                     // +32 bytes

                // Prefetch all 8 B fragments (two 16B chunks at k-offsets {0..7} and {16..23} (+hi*8))
                U bfr[8];
                const int baseHW = kk + hi * 8;      // halfword offset within row
#pragma unroll
                for (int ns = 0; ns < 8; ++ns) {
                    const uint4* xp4 = (const uint4*)(sX + (ns * 16 + row) * kc + baseHW);
                    bfr[ns].q[0] = xp4[0];
                    bfr[ns].q[1] = xp4[2];
                }
                SCHED_FENCE();   // keep all loads grouped ahead of the WMMAs
                // 8 back-to-back WMMAs, A reused from registers
#pragma unroll
                for (int ns = 0; ns < 8; ++ns) {
                    acc[ns] = __builtin_amdgcn_wmma_f32_16x16x32_bf16(
                        false, a.v, false, bfr[ns].v, (short)0, acc[ns], false, false);
                }
                SCHED_FENCE();   // keep next iteration's loads after this WMMA group
            }
        }
        __syncthreads();
    }

    if (active) {
#pragma unroll
        for (int ns = 0; ns < 8; ++ns) {
            float* yrow = Y + (size_t)(n0 + ns * 16 + row) * MP + mt * 16;
#pragma unroll
            for (int v = 0; v < 8; ++v) {
                const int ml = v + hi * 8;
                yrow[ml] = acc[ns][v] + bias[mt * 16 + ml];
            }
        }
    }
}

// ---------------------------------------------------------------------------
// Kernel 4: per-channel sum / sum-of-squares over Y [Nrows][MP].
// ---------------------------------------------------------------------------
__global__ void col_stats_kernel(const float* __restrict__ Y, float* __restrict__ stats,
                                 int MP, int Nrows) {
    __shared__ float s1[256], s2[256];
    const int c = blockIdx.x, t = threadIdx.x;
    float sum = 0.0f, sq = 0.0f;
    for (int q = t; q < Nrows; q += 256) {
        const float v = Y[(size_t)q * MP + c];
        sum += v; sq += v * v;
    }
    s1[t] = sum; s2[t] = sq;
    __syncthreads();
    for (int off = 128; off > 0; off >>= 1) {
        if (t < off) { s1[t] += s1[t + off]; s2[t] += s2[t + off]; }
        __syncthreads();
    }
    if (t == 0) { stats[c] = s1[0]; stats[MP + c] = s2[0]; }
}

// ---------------------------------------------------------------------------
// Kernel 5: training-mode BN + ReLU + f32->bf16 convert.
// ---------------------------------------------------------------------------
__global__ void bn_relu_cvt_kernel(const float* __restrict__ Y, const float* __restrict__ stats,
                                   const float* __restrict__ gamma, const float* __restrict__ beta,
                                   unsigned short* __restrict__ Xo, int MP, int Nrows) {
    const int idx = blockIdx.x * blockDim.x + threadIdx.x;
    if (idx >= Nrows * MP) return;
    const int c = idx % MP;
    const float inv  = 1.0f / (float)Nrows;
    const float mean = stats[c] * inv;
    const float var  = stats[MP + c] * inv - mean * mean;
    const float sc   = gamma[c] * rsqrtf(var + EPSBN);
    float v = (Y[idx] - mean) * sc + beta[c];
    v = fmaxf(v, 0.0f);
    Xo[idx] = f2bf(v);
}

// ---------------------------------------------------------------------------
// Kernel 6: max-pool over the 64 grid points. grid = BK blocks, 128 threads.
// ---------------------------------------------------------------------------
__global__ void maxpool_kernel(const unsigned short* __restrict__ X,
                               unsigned short* __restrict__ P) {
    const int bk = blockIdx.x, c = threadIdx.x;
    float m = -3.4e38f;
    for (int g = 0; g < GPv; ++g)
        m = fmaxf(m, bf2f(X[((size_t)bk * GPv + g) * MPc + c]));
    P[(size_t)bk * MPc + c] = f2bf(m);
}

// ---------------------------------------------------------------------------
// Kernel 7: slice the last 18 of 77 channels -> iou_scores [B,K,18]
// ---------------------------------------------------------------------------
__global__ void extract_kernel(const float* __restrict__ Y3, float* __restrict__ out) {
    const int idx = blockIdx.x * blockDim.x + threadIdx.x;
    if (idx >= BKv * 18) return;
    const int bk = idx / 18, j = idx % 18;
    out[idx] = Y3[(size_t)bk * MP3 + (OUT3v - 18) + j];
}

// ---------------------------------------------------------------------------
extern "C" void kernel_launch(void* const* d_in, const int* in_sizes, int n_in,
                              void* d_out, int out_size, void* d_ws, size_t ws_size,
                              hipStream_t stream) {
    (void)in_sizes; (void)n_in; (void)out_size; (void)ws_size;

    const float* center  = (const float*)d_in[0];
    const float* size_   = (const float*)d_in[1];
    const float* heading = (const float*)d_in[2];
    const float* sxyz    = (const float*)d_in[3];
    const float* sfeat   = (const float*)d_in[4];
    const float* w0 = (const float*)d_in[5],  *b0 = (const float*)d_in[6];
    const float* g0 = (const float*)d_in[7],  *be0 = (const float*)d_in[8];
    const float* w1 = (const float*)d_in[9],  *b1 = (const float*)d_in[10];
    const float* g1 = (const float*)d_in[11], *be1 = (const float*)d_in[12];
    const float* w2 = (const float*)d_in[13], *b2 = (const float*)d_in[14];
    const float* g2 = (const float*)d_in[15], *be2 = (const float*)d_in[16];
    const float* cw1 = (const float*)d_in[17], *cb1 = (const float*)d_in[18];
    const float* g3 = (const float*)d_in[19], *be3 = (const float*)d_in[20];
    const float* cw2 = (const float*)d_in[21], *cb2 = (const float*)d_in[22];
    const float* g4 = (const float*)d_in[23], *be4 = (const float*)d_in[24];
    const float* cw3 = (const float*)d_in[25], *cb3 = (const float*)d_in[26];
    float* out = (float*)d_out;

    // workspace carve-up
    uint8_t* ws = (uint8_t*)d_ws;
    size_t o = 0;
    auto bump = [&](size_t bytes) { size_t r = o; o += (bytes + 255) & ~(size_t)255; return r; };
    unsigned short* x0   = (unsigned short*)(ws + bump((size_t)NQv * CINP0 * 2)); // big bf16 buf
    float*          ybuf = (float*)         (ws + bump((size_t)NQv * MPc * 4));   // f32 GEMM out
    unsigned short* x1   = (unsigned short*)(ws + bump((size_t)NQv * MPc * 2));   // bf16 act buf
    unsigned short* w0b  = (unsigned short*)(ws + bump((size_t)MPc * CINP0 * 2));
    unsigned short* w1b  = (unsigned short*)(ws + bump((size_t)MPc * MPc * 2));
    unsigned short* w2b  = (unsigned short*)(ws + bump((size_t)MPc * MPc * 2));
    unsigned short* cw1b = (unsigned short*)(ws + bump((size_t)MPc * MPc * 2));
    unsigned short* cw2b = (unsigned short*)(ws + bump((size_t)MPc * MPc * 2));
    unsigned short* cw3b = (unsigned short*)(ws + bump((size_t)MP3 * MPc * 2));
    float*          cb3p = (float*)         (ws + bump((size_t)MP3 * 4));
    float*          stats= (float*)         (ws + bump((size_t)2 * MPc * 4));
    unsigned short* pool = (unsigned short*)(ws + bump((size_t)BKv * MPc * 2));
    unsigned short* fcA  = (unsigned short*)(ws + bump((size_t)BKv * MPc * 2));
    unsigned short* fcB  = (unsigned short*)(ws + bump((size_t)BKv * MPc * 2));

    // ---- weight / bias prep (bf16, padded) ----
    {
        int n = MPc * CINP0;
        cvt_w_kernel<<<(n + 255) / 256, 256, 0, stream>>>(w0, w0b, MPc, CIN0, MPc, CINP0);
        n = MPc * MPc;
        cvt_w_kernel<<<(n + 255) / 256, 256, 0, stream>>>(w1,  w1b,  MPc, MPc, MPc, MPc);
        cvt_w_kernel<<<(n + 255) / 256, 256, 0, stream>>>(w2,  w2b,  MPc, MPc, MPc, MPc);
        cvt_w_kernel<<<(n + 255) / 256, 256, 0, stream>>>(cw1, cw1b, MPc, MPc, MPc, MPc);
        cvt_w_kernel<<<(n + 255) / 256, 256, 0, stream>>>(cw2, cw2b, MPc, MPc, MPc, MPc);
        n = MP3 * MPc;
        cvt_w_kernel<<<(n + 255) / 256, 256, 0, stream>>>(cw3, cw3b, OUT3v, MPc, MP3, MPc);
        pad_bias_kernel<<<1, MP3, 0, stream>>>(cb3, cb3p, OUT3v, MP3);
    }

    // ---- stage 1: grid + 3NN + interpolation -> x0 [NQv][288] bf16 ----
    knn_interp_kernel<<<dim3(Kv, Bv), 256, 0, stream>>>(center, size_, heading, sxyz, sfeat, x0);

    const int bnBlocksBig = (NQv * MPc + 255) / 256;
    const int bnBlocksFC  = (BKv * MPc + 255) / 256;
    const int gemmBlocksBig = NQv / NBLK;   // 512
    const int gemmBlocksFC  = BKv / NBLK;   // 8

    // ---- conv layer 0: [128x288] @ x0 -> y; BN+ReLU -> x1 ----
    wmma_gemm_bf16_kernel<<<gemmBlocksBig, 256, 0, stream>>>(w0b, x0, b0, ybuf, MPc, CINP0);
    col_stats_kernel<<<MPc, 256, 0, stream>>>(ybuf, stats, MPc, NQv);
    bn_relu_cvt_kernel<<<bnBlocksBig, 256, 0, stream>>>(ybuf, stats, g0, be0, x1, MPc, NQv);

    // ---- conv layer 1: reuse x0 region as [NQv][128] bf16 output ----
    unsigned short* x0r = (unsigned short*)x0;
    wmma_gemm_bf16_kernel<<<gemmBlocksBig, 256, 0, stream>>>(w1b, x1, b1, ybuf, MPc, MPc);
    col_stats_kernel<<<MPc, 256, 0, stream>>>(ybuf, stats, MPc, NQv);
    bn_relu_cvt_kernel<<<bnBlocksBig, 256, 0, stream>>>(ybuf, stats, g1, be1, x0r, MPc, NQv);

    // ---- conv layer 2 ----
    wmma_gemm_bf16_kernel<<<gemmBlocksBig, 256, 0, stream>>>(w2b, x0r, b2, ybuf, MPc, MPc);
    col_stats_kernel<<<MPc, 256, 0, stream>>>(ybuf, stats, MPc, NQv);
    bn_relu_cvt_kernel<<<bnBlocksBig, 256, 0, stream>>>(ybuf, stats, g2, be2, x1, MPc, NQv);

    // ---- max-pool over 64 grid points -> pool [BK][128] bf16 ----
    maxpool_kernel<<<BKv, MPc, 0, stream>>>(x1, pool);

    // ---- FC layer 1 ----
    wmma_gemm_bf16_kernel<<<gemmBlocksFC, 256, 0, stream>>>(cw1b, pool, cb1, ybuf, MPc, MPc);
    col_stats_kernel<<<MPc, 256, 0, stream>>>(ybuf, stats, MPc, BKv);
    bn_relu_cvt_kernel<<<bnBlocksFC, 256, 0, stream>>>(ybuf, stats, g3, be3, fcA, MPc, BKv);

    // ---- FC layer 2 ----
    wmma_gemm_bf16_kernel<<<gemmBlocksFC, 256, 0, stream>>>(cw2b, fcA, cb2, ybuf, MPc, MPc);
    col_stats_kernel<<<MPc, 256, 0, stream>>>(ybuf, stats, MPc, BKv);
    bn_relu_cvt_kernel<<<bnBlocksFC, 256, 0, stream>>>(ybuf, stats, g4, be4, fcB, MPc, BKv);

    // ---- FC layer 3 (no BN) -> y3 [BK][80] f32, then slice ----
    wmma_gemm_bf16_kernel<<<gemmBlocksFC, 256, 0, stream>>>(cw3b, fcB, cb3p, ybuf, MP3, MPc);
    extract_kernel<<<(BKv * 18 + 255) / 256, 256, 0, stream>>>(ybuf, out);
}
